// MPNEncoder_12232066859189
// MI455X (gfx1250) — compile-verified, matrix operand order checked by reading, not run on hardware
//
#include <hip/hip_runtime.h>
#include <hip/hip_bf16.h>

typedef __attribute__((ext_vector_type(16))) __bf16 v16bf;
typedef __attribute__((ext_vector_type(8)))  __bf16 v8bf;
typedef __attribute__((ext_vector_type(8)))  float  v8f;

#define N_ATOMS   102400
#define N_BONDS   204800
#define MAX_NB    6
#define HIDDEN    256
#define ATOM_FDIM 133
#define BOND_FDIM 147
#define N_MOLS    4096
#define BOND_KPAD 160   // 147 padded to mult of 32
#define AMSG_OFF  144   // amsg region start in readout K-layout (mult of 8 -> aligned v8bf)
#define ATOM_KPAD 416   // 144 + 256 = 400 padded to mult of 32

#define LPITCH 40       // LDS pitch in halves: mult of 8 (16B align), spreads banks

// ---------------- elementwise helpers ----------------

// W [K][256] f32 -> Wt [256][KPAD] bf16 (zero-padded K)
__global__ void transpose_to_bf16(const float* __restrict__ W, __bf16* __restrict__ Wt,
                                  int K, int KPAD) {
    int idx = blockIdx.x * blockDim.x + threadIdx.x;
    if (idx >= HIDDEN * KPAD) return;
    int n = idx / KPAD, k = idx % KPAD;
    float v = (k < K) ? W[(size_t)k * HIDDEN + n] : 0.0f;
    Wt[idx] = (__bf16)v;
}

// W_o [389][256] f32 -> Wt [256][416] bf16 with K-gap layout:
//   k<133 -> W_o[k][n]; 144<=k<400 -> W_o[k-11][n]; else 0
__global__ void transpose_wo(const float* __restrict__ W, __bf16* __restrict__ Wt) {
    int idx = blockIdx.x * blockDim.x + threadIdx.x;
    if (idx >= HIDDEN * ATOM_KPAD) return;
    int n = idx / ATOM_KPAD, k = idx % ATOM_KPAD;
    float v = 0.0f;
    if (k < ATOM_FDIM)
        v = W[(size_t)k * HIDDEN + n];
    else if (k >= AMSG_OFF && k < AMSG_OFF + HIDDEN)
        v = W[(size_t)(k - (AMSG_OFF - ATOM_FDIM)) * HIDDEN + n];
    Wt[idx] = (__bf16)v;
}

// a_message[a][h] = sum_j message[a2b[a][j]][h]   (block = one atom row)
__global__ void neighbor_sum(const __bf16* __restrict__ msg, const int* __restrict__ a2b,
                             __bf16* __restrict__ amsg) {
    int a = blockIdx.x, h = threadIdx.x;
    float s = 0.0f;
#pragma unroll
    for (int j = 0; j < MAX_NB; ++j) {
        int b = a2b[a * MAX_NB + j];
        s += (float)msg[(size_t)b * HIDDEN + h];
    }
    amsg[(size_t)a * HIDDEN + h] = (__bf16)s;
}

__global__ void zero_f32(float* __restrict__ p, int n) {
    int i = blockIdx.x * blockDim.x + threadIdx.x;
    if (i < n) p[i] = 0.0f;
}

__global__ void count_atoms(const int* __restrict__ mol_ids, float* __restrict__ counts) {
    int i = blockIdx.x * blockDim.x + threadIdx.x;
    if (i < N_ATOMS) unsafeAtomicAdd(&counts[mol_ids[i]], 1.0f);
}

__global__ void normalize_out(float* __restrict__ out, const float* __restrict__ counts) {
    int i = blockIdx.x * blockDim.x + threadIdx.x;
    if (i >= N_MOLS * HIDDEN) return;
    int m = i / HIDDEN;
    out[i] = out[i] / fmaxf(counts[m], 1.0f);
}

// ---------------- WMMA GEMM ----------------
// ISA 16-bit A 16x32 layout: lane L (r=L&15, kb=(L>>4)*8) holds halves
//   [kb..kb+7] in VGPR0-3 and [kb+16..kb+23] in VGPR4-7 of row r.
// B stored transposed [N][KPAD] -> B-fragment load pattern identical over columns.
static __device__ inline v16bf load_frag(const __bf16* base) {
    v8bf lo = *(const v8bf*)(base);
    v8bf hi = *(const v8bf*)(base + 16);
    v16bf r;
#pragma unroll
    for (int i = 0; i < 8; ++i) { r[i] = lo[i]; r[i + 8] = hi[i]; }
    return r;
}

// Block tile 64(M) x 128(N), 8 waves (2x4), wave tile 32x32 = 2x2 WMMA frags.
//
// A-tile provider (fused into LDS staging; each thread fills 8 halves of one row).
// All boundary handling is branchless (clamped speculatable loads + selects) so
// the staging path stays a batch of parallel VMEM loads:
//   ASRC 1: bond combine  amsg[b2a[row]][k] - msgin[b2revb[row]][k]        (KPAD=256)
//   ASRC 2: atom concat   [f_atoms(133) | 0 -> 144 | amsg(256) | 0 -> 416]
//   ASRC 3: f32 source [M][BOND_FDIM] zero-padded, converted on the fly    (f_bonds)
//
// Epilogue:
//   EPI 0: out_f32 = acc; out_bf = relu(acc)             (inp GEMM)
//   EPI 1: out_bf  = relu(inpf[row,col] + acc)           (message update GEMM)
//   EPI 2: t = relu(acc + inpf[col]); atomicAdd sums[mol_ids[row]][col] += t
template <int ASRC, int EPI>
__global__ __launch_bounds__(256) void gemm_bf16(
    const __bf16* __restrict__ Abf,   // ASRC1: amsg; ASRC2: amsg
    const __bf16* __restrict__ Abf2,  // ASRC1: msg_in
    const float*  __restrict__ Af32,  // ASRC2: f_atoms; ASRC3: f_bonds
    const int*    __restrict__ idx0,  // ASRC1: b2a
    const int*    __restrict__ idx1,  // ASRC1: b2revb
    const __bf16* __restrict__ Bt, int KPAD,
    const float* __restrict__ inpf, float* __restrict__ out_f32,
    __bf16* __restrict__ out_bf, const int* __restrict__ mol_ids,
    float* __restrict__ sums) {
    __shared__ __bf16 sA[64 * LPITCH];
    __shared__ __bf16 sB[128 * LPITCH];

    const int tid  = threadIdx.x;
    const int lane = tid & 31;
    const int wave = tid >> 5;
    const int wm = wave >> 2;          // 0..1
    const int wn = wave & 3;           // 0..3
    const int m0 = blockIdx.x * 64;
    const int n0 = blockIdx.y * 128;

    const int r  = lane & 15;
    const int kb = (lane >> 4) << 3;   // 0 or 8

    // staging indices for this thread's A segment (row, 8-half segment)
    const int arow = tid >> 2;         // 0..63
    const int aseg = tid & 3;          // 0..3
    const int grow = m0 + arow;        // global A row

    v8f acc[2][2] = {};

    for (int k0 = 0; k0 < KPAD; k0 += 32) {
        const int kbase = k0 + aseg * 8;
        // ---- stage A: 64 rows x 32 halves, value computed per provider ----
        {
            v8bf av;
            if (ASRC == 1) {
                int a  = idx0[grow];   // b2a[b]
                int rb = idx1[grow];   // b2revb[b]
                v8bf x = *(const v8bf*)(Abf  + (size_t)a  * HIDDEN + kbase);
                v8bf y = *(const v8bf*)(Abf2 + (size_t)rb * HIDDEN + kbase);
#pragma unroll
                for (int i = 0; i < 8; ++i) av[i] = (__bf16)((float)x[i] - (float)y[i]);
            } else if (ASRC == 2) {
                // head: f_atoms with clamped unconditional loads; tail: aligned amsg v8bf
                if (kbase < AMSG_OFF) {
                    const float* rp = Af32 + (size_t)grow * ATOM_FDIM;
#pragma unroll
                    for (int i = 0; i < 8; ++i) {
                        int k = kbase + i;
                        float f = rp[(k < ATOM_FDIM) ? k : 0];   // speculatable load
                        av[i] = (__bf16)((k < ATOM_FDIM) ? f : 0.0f);
                    }
                } else {
                    int koff = kbase - AMSG_OFF;                  // mult of 8
                    int kc = (koff <= HIDDEN - 8) ? koff : (HIDDEN - 8);
                    v8bf x = *(const v8bf*)(Abf + (size_t)grow * HIDDEN + kc);
#pragma unroll
                    for (int i = 0; i < 8; ++i)
                        av[i] = (koff + i < HIDDEN) ? x[i] : (__bf16)0.0f;
                }
            } else { // ASRC == 3 : f32 [M][BOND_FDIM], pad to KPAD (branchless clamp)
                const float* rp = Af32 + (size_t)grow * BOND_FDIM;
#pragma unroll
                for (int i = 0; i < 8; ++i) {
                    int k = kbase + i;
                    float f = rp[(k < BOND_FDIM) ? k : 0];        // speculatable load
                    av[i] = (__bf16)((k < BOND_FDIM) ? f : 0.0f);
                }
            }
            *(v8bf*)(&sA[arow * LPITCH + aseg * 8]) = av;
        }
        // ---- stage B: 128 rows x 32 halves (thread: row=t>>1, 16-half segment) ----
        {
            int row = tid >> 1, seg = tid & 1;
            const v8bf* src = (const v8bf*)(Bt + (size_t)(n0 + row) * KPAD + k0 + seg * 16);
            *(v8bf*)(&sB[row * LPITCH + seg * 16])     = src[0];
            *(v8bf*)(&sB[row * LPITCH + seg * 16 + 8]) = src[1];
        }
        __syncthreads();

        v16bf af[2], bf[2];
#pragma unroll
        for (int fm = 0; fm < 2; ++fm)
            af[fm] = load_frag(&sA[(wm * 32 + fm * 16 + r) * LPITCH + kb]);
#pragma unroll
        for (int fn = 0; fn < 2; ++fn)
            bf[fn] = load_frag(&sB[(wn * 32 + fn * 16 + r) * LPITCH + kb]);

#pragma unroll
        for (int fm = 0; fm < 2; ++fm)
#pragma unroll
            for (int fn = 0; fn < 2; ++fn)
                acc[fm][fn] = __builtin_amdgcn_wmma_f32_16x16x32_bf16(
                    false, af[fm], false, bf[fn], (short)0, acc[fm][fn], false, false);
        __syncthreads();
    }

    // epilogue; C/D layout: lane holds col n=lane&15; VGPR v -> row v (+8 if lane>=16)
    const int mshift = (lane >> 4) << 3;
#pragma unroll
    for (int fm = 0; fm < 2; ++fm) {
#pragma unroll
        for (int fn = 0; fn < 2; ++fn) {
            int ncol = n0 + wn * 32 + fn * 16 + (lane & 15);
            int mbase = m0 + wm * 32 + fm * 16 + mshift;
#pragma unroll
            for (int v = 0; v < 8; ++v) {
                int mrow = mbase + v;
                float val = acc[fm][fn][v];
                size_t off = (size_t)mrow * HIDDEN + ncol;
                if (EPI == 0) {
                    out_f32[off] = val;
                    out_bf[off]  = (__bf16)fmaxf(val, 0.0f);
                } else if (EPI == 1) {
                    float t = inpf[off] + val;
                    out_bf[off] = (__bf16)fmaxf(t, 0.0f);
                } else {
                    float t = fmaxf(val + inpf[ncol], 0.0f);
                    unsafeAtomicAdd(&sums[(size_t)mol_ids[mrow] * HIDDEN + ncol], t);
                }
            }
        }
    }
}

// ---------------- launch ----------------
extern "C" void kernel_launch(void* const* d_in, const int* in_sizes, int n_in,
                              void* d_out, int out_size, void* d_ws, size_t ws_size,
                              hipStream_t stream) {
    const float* f_atoms = (const float*)d_in[0];
    const float* f_bonds = (const float*)d_in[1];
    const float* W_i     = (const float*)d_in[2];
    const float* W_h1    = (const float*)d_in[3];
    const float* W_h2    = (const float*)d_in[4];
    const float* W_o     = (const float*)d_in[5];
    const float* b_o     = (const float*)d_in[6];
    const int*   a2b     = (const int*)d_in[7];
    const int*   b2a     = (const int*)d_in[8];
    const int*   b2revb  = (const int*)d_in[9];
    const int*   mol_ids = (const int*)d_in[10];
    float* out = (float*)d_out;   // [N_MOLS][HIDDEN], used as segment-sum accumulator

    // workspace layout (~472 MB)
    char* w = (char*)d_ws;
    float*  inp    = (float*)w;  w += (size_t)N_BONDS * HIDDEN * 4;   // fp32 residual
    __bf16* msgA   = (__bf16*)w; w += (size_t)N_BONDS * HIDDEN * 2;
    __bf16* msgB   = (__bf16*)w; w += (size_t)N_BONDS * HIDDEN * 2;
    __bf16* amsg   = (__bf16*)w; w += (size_t)N_ATOMS * HIDDEN * 2;
    __bf16* Wi_t   = (__bf16*)w; w += (size_t)HIDDEN * BOND_KPAD * 2;
    __bf16* Wh1_t  = (__bf16*)w; w += (size_t)HIDDEN * HIDDEN * 2;
    __bf16* Wh2_t  = (__bf16*)w; w += (size_t)HIDDEN * HIDDEN * 2;
    __bf16* Wo_t   = (__bf16*)w; w += (size_t)HIDDEN * ATOM_KPAD * 2;
    float*  counts = (float*)w;  w += (size_t)N_MOLS * 4;

    // 1) weights -> transposed bf16 (W_o with the K-gap layout)
    transpose_to_bf16<<<(HIDDEN * BOND_KPAD + 255) / 256, 256, 0, stream>>>(W_i,  Wi_t,  BOND_FDIM, BOND_KPAD);
    transpose_to_bf16<<<(HIDDEN * HIDDEN   + 255) / 256, 256, 0, stream>>>(W_h1, Wh1_t, HIDDEN, HIDDEN);
    transpose_to_bf16<<<(HIDDEN * HIDDEN   + 255) / 256, 256, 0, stream>>>(W_h2, Wh2_t, HIDDEN, HIDDEN);
    transpose_wo<<<(HIDDEN * ATOM_KPAD + 255) / 256, 256, 0, stream>>>(W_o, Wo_t);

    // 2) inp = f_bonds @ W_i ; msgA = relu(inp)   (f32->bf16 pad fused into A staging)
    gemm_bf16<3, 0><<<dim3(N_BONDS / 64, HIDDEN / 128), 256, 0, stream>>>(
        nullptr, nullptr, f_bonds, nullptr, nullptr,
        Wi_t, BOND_KPAD, nullptr, inp, msgA, nullptr, nullptr);

    // 3) two message-passing steps (bond combine fused into A staging)
    for (int d = 0; d < 2; ++d) {
        __bf16* min_  = d ? msgB : msgA;
        __bf16* mout_ = d ? msgA : msgB;
        const __bf16* Wh = d ? Wh2_t : Wh1_t;
        neighbor_sum<<<N_ATOMS, HIDDEN, 0, stream>>>(min_, a2b, amsg);
        gemm_bf16<1, 1><<<dim3(N_BONDS / 64, HIDDEN / 128), 256, 0, stream>>>(
            amsg, min_, nullptr, b2a, b2revb,
            Wh, HIDDEN, inp, nullptr, mout_, nullptr, nullptr);
    }
    // final message lives in msgA

    // 4) final neighbor sum
    neighbor_sum<<<N_ATOMS, HIDDEN, 0, stream>>>(msgA, a2b, amsg);

    // 5) zero accumulators (d_out is poisoned by harness), counts
    zero_f32<<<(N_MOLS * HIDDEN + 255) / 256, 256, 0, stream>>>(out, N_MOLS * HIDDEN);
    zero_f32<<<(N_MOLS + 255) / 256, 256, 0, stream>>>(counts, N_MOLS);
    count_atoms<<<(N_ATOMS + 255) / 256, 256, 0, stream>>>(mol_ids, counts);

    // 6) readout GEMM: concat(f_atoms, amsg) fused into A staging; relu(+b_o);
    //    fused per-molecule segment-sum via fp32 global atomics into d_out
    gemm_bf16<2, 2><<<dim3(N_ATOMS / 64, HIDDEN / 128), 256, 0, stream>>>(
        amsg, nullptr, f_atoms, nullptr, nullptr,
        Wo_t, ATOM_KPAD, b_o, nullptr, nullptr, mol_ids, out);

    // 7) mean
    normalize_out<<<(N_MOLS * HIDDEN + 255) / 256, 256, 0, stream>>>(out, counts);
}
